// AWD_LSTM_28295244546434
// MI455X (gfx1250) — matrix operand order repaired and checked by reading, not compiled, not run on hardware
//
#include <hip/hip_runtime.h>
#include <hip/hip_bf16.h>

// ---------------------------------------------------------------------------
// AWD-LSTM forward on MI455X (gfx1250), wave32 + WMMA bf16 + TDM tile loads.
//  B=32, T=128, V=32000, E=400, M=512, H=1152.
//  - GEMM tiles staged into LDS by the Tensor Data Mover (tensor_load_to_lds,
//    s_wait_tensorcnt), with LDS row padding applied by the TDM pad feature.
//  - All K dims padded to multiples of 32 (E: 400->416) so staging and
//    fragment loads are pure b128 vectors.
//  - Persistent recurrent scan keeps cell state in WMMA accumulators.
// ---------------------------------------------------------------------------

#define BB 32
#define TT 128
#define VV 32000
#define EE 400
#define EP 416      // E padded to /32
#define MM 512
#define HH 1152

typedef __attribute__((ext_vector_type(16))) __bf16 v16bf;
typedef __attribute__((ext_vector_type(8)))  float  v8f;
typedef __attribute__((ext_vector_type(4)))  unsigned int u32x4;
typedef __attribute__((ext_vector_type(4)))  int i32x4;
typedef __attribute__((ext_vector_type(8)))  int i32x8;

union Frag16 {
    v16bf v;
    unsigned short u[16];
    uint4 q[2];
};

__device__ __forceinline__ unsigned short f2bf(float f) {
    unsigned int u = __float_as_uint(f);
    unsigned int r = (u + 0x7FFFu + ((u >> 16) & 1u)) >> 16;   // RNE
    return (unsigned short)r;
}

__device__ __forceinline__ v8f wmma_bf16(v16bf a, v16bf b, v8f c) {
    return __builtin_amdgcn_wmma_f32_16x16x32_bf16(
        false, a, false, b, (short)0, c, false, false);
}

__device__ __forceinline__ float sigmf(float x) {
    return 1.0f / (1.0f + __expf(-x));
}

// ---------------------------------------------------------------------------
// TDM: load a 2-D tile (rows x 32 bf16) from a row-major matrix into LDS.
// LDS rows are padded 32 -> 40 elements by the TDM pad unit
// (pad_interval = 16 DWORDs, pad_amount = 4 DWORDs).
// Descriptor layout per CDNA5 ISA (D# groups 0/1).
// 6-arg builtin form (amdgpu-toolchain / clang-23).
// ---------------------------------------------------------------------------
__device__ __forceinline__ void tdm_load_tile32(unsigned lds_off,
                                                const unsigned short* gsrc,
                                                int rows, int rowStrideElems) {
    unsigned long long ga = (unsigned long long)(uintptr_t)gsrc;
    u32x4 g0;
    g0[0] = 1u;                                           // count=1, user mode
    g0[1] = lds_off;                                      // lds_addr (bytes)
    g0[2] = (unsigned)(ga & 0xffffffffu);                 // global_addr[31:0]
    g0[3] = (unsigned)((ga >> 32) & 0x01ffffffu)          // global_addr[56:32]
          | (2u << 30);                                   // type=2 (image)
    i32x8 g1;
    g1[0] = (1 << 16)        // data_size = 1 -> 2 bytes
          | (1 << 20)        // pad_enable
          | (3 << 22)        // pad_interval: 16 DWORDs (one 32-elem bf16 row)
          | (3 << 25);       // pad_amount:   4 DWORDs (8 bf16 elements)
    g1[1] = (int)(32u << 16);                 // tensor_dim0[15:0]  (=32 elems)
    g1[2] = (int)(((unsigned)rows & 0xffffu) << 16);      // tensor_dim1 lo16
    g1[3] = (int)(32u << 16);                 // tile_dim0 = 32 (bits 127:112)
    g1[4] = rows & 0xffff;                    // tile_dim1 = rows, tile_dim2=0
    g1[5] = rowStrideElems;                   // tensor_dim0_stride[31:0]
    g1[6] = 0;
    g1[7] = 0;
    i32x4 z4 = {0, 0, 0, 0};
    i32x8 z8 = {0, 0, 0, 0, 0, 0, 0, 0};
    __builtin_amdgcn_tensor_load_to_lds(g0, g1, z4, z4, z8, 0);
}

// ---------------------------------------------------------------------------
// f32 -> bf16 convert with row zero-padding K -> Kp.
// ---------------------------------------------------------------------------
__global__ void cvt_pad_kernel(const float* __restrict__ src,
                               unsigned short* __restrict__ dst,
                               int K, int Kp, long n) {
    long i = (long)blockIdx.x * blockDim.x + threadIdx.x;
    if (i >= n) return;
    long r = i / Kp;
    int  k = (int)(i - r * Kp);
    dst[i] = (k < K) ? f2bf(src[r * K + k]) : (unsigned short)0;
}

// ---------------------------------------------------------------------------
// Tiled WMMA GEMM:  C[M,N] = A[M,K] * B + bias
//   BT=true : B stored [N][K] row-major -> B tile staged by TDM as [n][k]
//   BT=false: B stored [K][N] row-major -> manual staging as [k][n]
//   GATHER  : A row r is A[ids[r]] -> manual vectorized staging
// Requires: K % 32 == 0, M % 128 == 0, N % 64 == 0 (buffers pre-padded).
// Block = 256 threads (8 waves), tile 128(M) x 64(N), K-step 32.
// ---------------------------------------------------------------------------
template <bool BT, bool GATHER>
__global__ __launch_bounds__(256) void gemm_bf16_kernel(
    const unsigned short* __restrict__ A, int lda,
    const int* __restrict__ ids,
    const unsigned short* __restrict__ B, int ldb,
    const float* __restrict__ bias0, const float* __restrict__ bias1,
    float* __restrict__ Cf, long ldc,
    unsigned short* __restrict__ Cb, long ldcb,
    int M, int N, int K) {

    // A: [128 rows][40] (32 used + 8 pad). B: BT -> [64 n][40 k] ; !BT -> [32 k][72 n]
    __shared__ alignas(16) unsigned short As[128 * 40];
    __shared__ alignas(16) unsigned short Bs[64 * 40];

    const int tid  = threadIdx.x;
    const int wave = tid >> 5;
    const int lane = tid & 31;
    const int wm   = wave >> 1;      // 0..3
    const int wn   = wave & 1;       // 0..1
    const int half = lane >> 4;
    const int lm   = lane & 15;

    const unsigned As_off = (unsigned)(size_t)(&As[0]);
    const unsigned Bs_off = (unsigned)(size_t)(&Bs[0]);

    v8f vzero = {0.f, 0.f, 0.f, 0.f, 0.f, 0.f, 0.f, 0.f};
    v8f acc[2][2];
    for (int i = 0; i < 2; ++i)
        for (int j = 0; j < 2; ++j) acc[i][j] = vzero;

    for (int k0 = 0; k0 < K; k0 += 32) {
        __syncthreads();

        // ---- stage A tile 128x32 ----
        if constexpr (GATHER) {
            int row = tid >> 1, chunk = tid & 1;          // 256 thr = 128x2
            long ar = (long)ids[blockIdx.y * 128 + row];
            const uint4* src = (const uint4*)(A + ar * lda + k0 + chunk * 16);
            uint4 v0 = src[0];
            uint4 v1 = src[1];
            if (k0 + 32 < K) __builtin_prefetch(A + ar * lda + k0 + 32, 0, 3);
            *(uint4*)&As[row * 40 + chunk * 16]     = v0;
            *(uint4*)&As[row * 40 + chunk * 16 + 8] = v1;
        } else {
            if (tid < 32)
                tdm_load_tile32(As_off, A + (long)(blockIdx.y * 128) * lda + k0,
                                128, lda);
        }

        // ---- stage B tile ----
        if constexpr (BT) {
            if (tid < 32)
                tdm_load_tile32(Bs_off, B + (long)(blockIdx.x * 64) * ldb + k0,
                                64, ldb);
        } else {
            for (int i = tid; i < 64 * 32; i += 256) {
                int n = i & 63, kk = i >> 6;
                Bs[kk * 72 + n] = B[(long)(k0 + kk) * ldb + blockIdx.x * 64 + n];
            }
        }

        if constexpr (BT || !GATHER) {
            if (tid < 32) __builtin_amdgcn_s_wait_tensorcnt(0);
        }
        __syncthreads();

        // ---- fragments (A, B: element s of lane half h is k = 16h + s) ----
        Frag16 a[2], b[2];
        #pragma unroll
        for (int mt = 0; mt < 2; ++mt) {
            int row = 32 * wm + 16 * mt + lm;
            a[mt].q[0] = *(const uint4*)&As[row * 40 + half * 16];
            a[mt].q[1] = *(const uint4*)&As[row * 40 + half * 16 + 8];
        }
        #pragma unroll
        for (int nt = 0; nt < 2; ++nt) {
            int col = 32 * wn + 16 * nt + lm;
            if constexpr (BT) {
                b[nt].q[0] = *(const uint4*)&Bs[col * 40 + half * 16];
                b[nt].q[1] = *(const uint4*)&Bs[col * 40 + half * 16 + 8];
            } else {
                #pragma unroll
                for (int j = 0; j < 16; ++j)
                    b[nt].u[j] = Bs[(half * 16 + j) * 72 + col];
            }
        }
        #pragma unroll
        for (int mt = 0; mt < 2; ++mt)
            #pragma unroll
            for (int nt = 0; nt < 2; ++nt)
                acc[mt][nt] = wmma_bf16(a[mt].v, b[nt].v, acc[mt][nt]);
    }

    // ---- epilogue: D layout row m = r + 8*half, col n = lane&15 ----
    for (int mt = 0; mt < 2; ++mt) {
        for (int nt = 0; nt < 2; ++nt) {
            int gn = blockIdx.x * 64 + 32 * wn + 16 * nt + lm;
            float bb = (bias0 ? bias0[gn] : 0.f) + (bias1 ? bias1[gn] : 0.f);
            #pragma unroll
            for (int r = 0; r < 8; ++r) {
                int gm = blockIdx.y * 128 + 32 * wm + 16 * mt + r + 8 * half;
                float v = acc[mt][nt][r] + bb;
                if (Cf) Cf[(long)gm * ldc + gn] = v;
                if (Cb) Cb[(long)gm * ldcb + gn] = f2bf(v);
            }
        }
    }
    (void)M; (void)N;
}

// ---------------------------------------------------------------------------
// Grid-wide barrier for the persistent scan kernel (<=18 co-resident blocks).
// counter zeroed by hipMemsetAsync before each launch.
// ---------------------------------------------------------------------------
__device__ __forceinline__ void grid_bar(int* counter, int nblocks, int phase) {
    __threadfence();
    __syncthreads();
    if (threadIdx.x == 0) {
        __hip_atomic_fetch_add(counter, 1, __ATOMIC_RELEASE, __HIP_MEMORY_SCOPE_AGENT);
        const int target = nblocks * phase;
        while (__hip_atomic_load(counter, __ATOMIC_ACQUIRE, __HIP_MEMORY_SCOPE_AGENT) < target)
            __builtin_amdgcn_s_sleep(1);
    }
    __syncthreads();
    __threadfence();
}

// ---------------------------------------------------------------------------
// Persistent LSTM recurrent scan.
//  pre : [B*T][4H] f32, Whh: [4H][Kp] bf16 (K padded), hbuf: [B][Kp] bf16.
//  Each wave owns one 16(batch) x 16(hidden) tile and all 4 of its gate
//  tiles; cell state c is register-resident (v8f) across all 128 steps.
// ---------------------------------------------------------------------------
__global__ __launch_bounds__(256) void lstm_scan_kernel(
    const float* __restrict__ pre,
    const unsigned short* __restrict__ Whh,
    unsigned short* __restrict__ hbuf,
    const float* __restrict__ h0, const float* __restrict__ c0,
    unsigned short* __restrict__ y, int ldy,         // [B*T][ldy] bf16
    float* __restrict__ hN, float* __restrict__ cN,  // [B][H] f32
    int H, int Kp, int* counter, int nblocks, int units) {

    const int tid  = threadIdx.x;
    const int lane = tid & 31;
    const int u    = blockIdx.x * 8 + (tid >> 5);
    const bool active = (u < units);
    const int mt   = u & 1;
    const int nt   = u >> 1;
    const int half = lane >> 4;
    const int lm   = lane & 15;
    const int nn   = 16 * nt + lm;
    const int brow = 16 * mt + lm;
    const int ksteps = Kp >> 5;
    const int fourH = 4 * H;

    v8f vzero = {0.f, 0.f, 0.f, 0.f, 0.f, 0.f, 0.f, 0.f};
    v8f creg  = vzero;

    if (active) {
        #pragma unroll
        for (int r = 0; r < 8; ++r) {
            int b = 16 * mt + r + 8 * half;
            creg[r] = c0[b * H + nn];
            hbuf[(long)b * Kp + nn] = f2bf(h0[b * H + nn]);
        }
    }
    grid_bar(counter, nblocks, 1);
    int phase = 1;

    for (int t = 0; t < TT; ++t) {
        if (active) {
            v8f g[4];
            #pragma unroll
            for (int gi = 0; gi < 4; ++gi) g[gi] = vzero;
            #pragma unroll
            for (int r = 0; r < 8; ++r) {
                long row = (long)(16 * mt + r + 8 * half) * TT + t;
                const float* p = pre + row * fourH;
                g[0][r] = p[0 * H + nn];
                g[1][r] = p[1 * H + nn];
                g[2][r] = p[2 * H + nn];
                g[3][r] = p[3 * H + nn];
            }
            for (int s = 0; s < ksteps; ++s) {
                Frag16 a;
                const unsigned short* ap = hbuf + (long)brow * Kp + 32 * s + half * 16;
                a.q[0] = *(const uint4*)(ap);
                a.q[1] = *(const uint4*)(ap + 8);
                #pragma unroll
                for (int gi = 0; gi < 4; ++gi) {
                    const unsigned short* bp =
                        Whh + (long)(gi * H + nn) * Kp + 32 * s + half * 16;
                    Frag16 bf;
                    bf.q[0] = *(const uint4*)(bp);
                    bf.q[1] = *(const uint4*)(bp + 8);
                    g[gi] = wmma_bf16(a.v, bf.v, g[gi]);
                }
            }
            #pragma unroll
            for (int r = 0; r < 8; ++r) {
                float iv = sigmf(g[0][r]);
                float fv = sigmf(g[1][r]);
                float gv = tanhf(g[2][r]);
                float ov = sigmf(g[3][r]);
                float c  = fv * creg[r] + iv * gv;
                float h  = ov * tanhf(c);
                creg[r]  = c;
                int b = 16 * mt + r + 8 * half;
                unsigned short hb = f2bf(h);
                hbuf[(long)b * Kp + nn]          = hb;
                y[((long)b * TT + t) * ldy + nn] = hb;
                if (t == TT - 1) {
                    hN[b * H + nn] = h;
                    cN[b * H + nn] = c;
                }
            }
        }
        ++phase;
        grid_bar(counter, nblocks, phase);
    }
}

// ---------------------------------------------------------------------------
// Host-side orchestration.
// ---------------------------------------------------------------------------
extern "C" void kernel_launch(void* const* d_in, const int* in_sizes, int n_in,
                              void* d_out, int out_size, void* d_ws, size_t ws_size,
                              hipStream_t stream) {
    (void)in_sizes; (void)n_in; (void)out_size; (void)ws_size;

    const int*   ids   = (const int*)  d_in[0];
    const float* emb   = (const float*)d_in[1];
    const float* defW  = (const float*)d_in[2];
    const float* defb  = (const float*)d_in[3];
    const float* Wih[3] = {(const float*)d_in[4],  (const float*)d_in[10], (const float*)d_in[16]};
    const float* Whh[3] = {(const float*)d_in[5],  (const float*)d_in[11], (const float*)d_in[17]};
    const float* bih[3] = {(const float*)d_in[6],  (const float*)d_in[12], (const float*)d_in[18]};
    const float* bhh[3] = {(const float*)d_in[7],  (const float*)d_in[13], (const float*)d_in[19]};
    const float* h0[3]  = {(const float*)d_in[8],  (const float*)d_in[14], (const float*)d_in[20]};
    const float* c0[3]  = {(const float*)d_in[9],  (const float*)d_in[15], (const float*)d_in[21]};

    float* out = (float*)d_out;
    const long OFF_H0 = (long)BB * TT * VV;
    const long OFF_C0 = OFF_H0 + (long)BB * HH;
    const long OFF_H1 = OFF_C0 + (long)BB * HH;
    const long OFF_C1 = OFF_H1 + (long)BB * HH;
    const long OFF_H2 = OFF_C1 + (long)BB * HH;
    const long OFF_C2 = OFF_H2 + (long)BB * EE;
    (void)OFF_C2;

    size_t off = 0;
    auto alloc = [&](size_t bytes) -> void* {
        off = (off + 255) & ~(size_t)255;
        void* p = (char*)d_ws + off;
        off += bytes;
        return p;
    };
    unsigned short* emb_bf  = (unsigned short*)alloc((size_t)VV * EP * 2);   // [V][416]
    unsigned short* defW_bf = (unsigned short*)alloc((size_t)EP * MM * 2);   // [416][512]
    unsigned short* Wih_bf[3] = {
        (unsigned short*)alloc((size_t)4 * HH * MM * 2),
        (unsigned short*)alloc((size_t)4 * HH * HH * 2),
        (unsigned short*)alloc((size_t)4 * EE * HH * 2)};
    unsigned short* Whh_bf[3] = {
        (unsigned short*)alloc((size_t)4 * HH * HH * 2),
        (unsigned short*)alloc((size_t)4 * HH * HH * 2),
        (unsigned short*)alloc((size_t)4 * EE * EP * 2)};
    unsigned short* x0  = (unsigned short*)alloc((size_t)BB * TT * MM * 2);
    unsigned short* y0  = (unsigned short*)alloc((size_t)BB * TT * HH * 2);
    unsigned short* y1  = (unsigned short*)alloc((size_t)BB * TT * HH * 2);
    unsigned short* y2  = (unsigned short*)alloc((size_t)BB * TT * EP * 2);  // [4096][416]
    float*          pre = (float*)alloc((size_t)BB * TT * 4 * HH * 4);
    unsigned short* hbuf = (unsigned short*)alloc((size_t)BB * HH * 2);
    int*            counter = (int*)alloc(256);

    // zero-fill pads (defW rows 400..415, y2 cols 400..415)
    (void)hipMemsetAsync(defW_bf, 0, (size_t)EP * MM * 2, stream);
    (void)hipMemsetAsync(y2,      0, (size_t)BB * TT * EP * 2, stream);

    auto cvt = [&](const float* src, unsigned short* dst, long rows, int K, int Kp) {
        long n = rows * Kp;
        cvt_pad_kernel<<<dim3((unsigned)((n + 255) / 256)), 256, 0, stream>>>(src, dst, K, Kp, n);
    };

    cvt(emb,    emb_bf,  VV,     EE, EP);     // column-padded
    cvt(defW,   defW_bf, EE,     MM, MM);     // rows 400..415 stay zero
    cvt(Wih[0], Wih_bf[0], 4 * HH, MM, MM);
    cvt(Wih[1], Wih_bf[1], 4 * HH, HH, HH);
    cvt(Wih[2], Wih_bf[2], 4 * EE, HH, HH);
    cvt(Whh[0], Whh_bf[0], 4 * HH, HH, HH);
    cvt(Whh[1], Whh_bf[1], 4 * HH, HH, HH);
    cvt(Whh[2], Whh_bf[2], 4 * EE, EE, EP);

    const int Mrows = BB * TT;                 // 4096
    dim3 blk(256);

    // ---- embedding gather + DeFINE: x0 = emb[ids] @ defW + b (bf16) ----
    gemm_bf16_kernel<false, true><<<dim3(MM / 64, Mrows / 128), blk, 0, stream>>>(
        emb_bf, EP, ids, defW_bf, MM, defb, nullptr,
        nullptr, 0, x0, MM, Mrows, MM, EP);

    const unsigned short* xin[3] = {x0, y0, y1};
    const int kin[3]  = {MM, HH, HH};
    const int hdim[3] = {HH, HH, EE};
    const int kp[3]   = {HH, HH, EP};
    const int ldy[3]  = {HH, HH, EP};
    unsigned short* yout[3] = {y0, y1, y2};
    float* hNp[3] = {out + OFF_H0, out + OFF_H1, out + OFF_H2};
    float* cNp[3] = {out + OFF_C0, out + OFF_C1, out + OFF_C2};

    for (int l = 0; l < 3; ++l) {
        const int H = hdim[l];
        gemm_bf16_kernel<true, false><<<dim3((4 * H) / 64, Mrows / 128), blk, 0, stream>>>(
            xin[l], kin[l], nullptr, Wih_bf[l], kin[l], bih[l], bhh[l],
            pre, 4 * H, nullptr, 0, Mrows, 4 * H, kin[l]);

        (void)hipMemsetAsync(hbuf, 0, (size_t)BB * HH * 2, stream);
        (void)hipMemsetAsync(counter, 0, sizeof(int), stream);

        const int units = 2 * (H / 16);        // 144 / 144 / 50
        const int nblk  = (units + 7) / 8;     // 18 / 18 / 7
        lstm_scan_kernel<<<dim3(nblk), blk, 0, stream>>>(
            pre, Whh_bf[l], hbuf, h0[l], c0[l], yout[l], ldy[l],
            hNp[l], cNp[l], H, kp[l], counter, nblk, units);
    }

    // ---- tied-embedding logits: [4096,416] @ emb_bf^T -> [4096,32000] ----
    gemm_bf16_kernel<true, false><<<dim3(VV / 64, Mrows / 128), blk, 0, stream>>>(
        y2, EP, nullptr, emb_bf, EP, nullptr, nullptr,
        out, VV, nullptr, 0, Mrows, VV, EP);
}
